// AttentionAggregator_3195455668651
// MI455X (gfx1250) — matrix operand order repaired and checked by reading, not compiled, not run on hardware
//
#include <hip/hip_runtime.h>

typedef __attribute__((ext_vector_type(16))) _Float16 v16h;
typedef __attribute__((ext_vector_type(8)))  float    v8f;

#define D 128
#define H 64

// Monotone float <-> uint encoding so unsigned atomicMax == float max.
__device__ __forceinline__ unsigned fenc(float f) {
    unsigned u = __float_as_uint(f);
    return (u & 0x80000000u) ? ~u : (u | 0x80000000u);
}
__device__ __forceinline__ float fdec(unsigned u) {
    return (u & 0x80000000u) ? __uint_as_float(u & 0x7FFFFFFFu)
                             : __uint_as_float(~u);
}

// Native L2 float atomic add (global_atomic_add_f32).
__device__ __forceinline__ void atomicAddF32(float* p, float v) {
    unsafeAtomicAdd(p, v);
}

// Hardware v_tanh_f32 when available (gfx1250 TRANS op), else ocml.
__device__ __forceinline__ float fast_tanh(float x) {
#if __has_builtin(__builtin_amdgcn_tanhf)
    return __builtin_amdgcn_tanhf(x);
#else
    return tanhf(x);
#endif
}

// ---------------------------------------------------------------- init
__global__ void init_k(float* __restrict__ out, float* __restrict__ segsum,
                       unsigned* __restrict__ segmax, int out_size, int dim) {
    int i = blockIdx.x * blockDim.x + threadIdx.x;
    if (i < out_size) out[i] = 0.0f;
    if (i < dim) { segsum[i] = 0.0f; segmax[i] = 0u; }
}

// ------------------------------------------------- pre-pack W1 B-frags
// B fragment layout for v_wmma_f32_16x16x32_f16 (32x16 16-bit B):
// lane = hi*16+lo, element j of v16h holds B[K = kt*32 + (j<8 ? hi*8+j : 16+hi*8+j-8)][N = nt*16 + lo].
__global__ void bprep_k(const float* __restrict__ W1, _Float16* __restrict__ bfrag) {
    int tid  = threadIdx.x;          // 0..511
    int frag = tid >> 5;             // nt*4 + kt
    int lane = tid & 31;
    int nt = frag >> 2, kt = frag & 3;
    int lo = lane & 15, hi = lane >> 4;
    _Float16* dst = bfrag + ((size_t)frag * 32 + lane) * 16;
#pragma unroll
    for (int j = 0; j < 16; ++j) {
        int k = kt * 32 + (j < 8 ? hi * 8 + j : 16 + hi * 8 + (j - 8));
        int n = nt * 16 + lo;
        dst[j] = (_Float16)W1[k * H + n];
    }
}

// -------------------------------------------------------- score pass
// One wave computes s for 16 messages: tanh(msg@W1 + b1)@W2 + b2.
__global__ __launch_bounds__(256) void score_k(
    const float* __restrict__ msg, const int* __restrict__ index,
    const float* __restrict__ b1, const float* __restrict__ W2,
    const float* __restrict__ b2, const _Float16* __restrict__ bfrag,
    float* __restrict__ s_out, unsigned* __restrict__ segmax, int n_msgs) {

    int wave = threadIdx.x >> 5;
    int lane = threadIdx.x & 31;
    int m0 = (blockIdx.x * 8 + wave) * 16;
    if (m0 >= n_msgs) return;                      // wave-uniform exit
    int lo = lane & 15, hi = lane >> 4;

    // A fragments: lane element j holds msg[M = m0+lo][K = kt*32 + (j<8 ? hi*8+j : 16+hi*8+j-8)]
    const float4* row = (const float4*)(msg + (size_t)(m0 + lo) * D);
    v16h a[4];
#pragma unroll
    for (int kt = 0; kt < 4; ++kt) {
        const float4* rp = row + kt * 8 + hi * 2;
        float4 p0 = rp[0], p1 = rp[1], p2 = rp[4], p3 = rp[5];
        a[kt][0]  = (_Float16)p0.x; a[kt][1]  = (_Float16)p0.y;
        a[kt][2]  = (_Float16)p0.z; a[kt][3]  = (_Float16)p0.w;
        a[kt][4]  = (_Float16)p1.x; a[kt][5]  = (_Float16)p1.y;
        a[kt][6]  = (_Float16)p1.z; a[kt][7]  = (_Float16)p1.w;
        a[kt][8]  = (_Float16)p2.x; a[kt][9]  = (_Float16)p2.y;
        a[kt][10] = (_Float16)p2.z; a[kt][11] = (_Float16)p2.w;
        a[kt][12] = (_Float16)p3.x; a[kt][13] = (_Float16)p3.y;
        a[kt][14] = (_Float16)p3.z; a[kt][15] = (_Float16)p3.w;
    }

    float sacc[8] = {0.f, 0.f, 0.f, 0.f, 0.f, 0.f, 0.f, 0.f};
    const v16h* bfp = (const v16h*)bfrag;

#pragma unroll
    for (int nt = 0; nt < 4; ++nt) {
        float b1v = b1[nt * 16 + lo];
        float w2v = W2[nt * 16 + lo];
        v8f c = {};
#pragma unroll
        for (int kt = 0; kt < 4; ++kt) {
            v16h b = bfp[(nt * 4 + kt) * 32 + lane];
            c = __builtin_amdgcn_wmma_f32_16x16x32_f16(
                    false, a[kt], false, b, (short)0, c, false, false);
        }
        // C layout: VGPR r, lane(hi,lo): element (M = r + 8*hi, N = nt*16 + lo)
#pragma unroll
        for (int r = 0; r < 8; ++r)
            sacc[r] += fast_tanh(c[r] + b1v) * w2v;
    }

    // Reduce over the 16 lanes of each half (N dimension).
#pragma unroll
    for (int r = 0; r < 8; ++r) {
        sacc[r] += __shfl_xor(sacc[r], 1, 32);
        sacc[r] += __shfl_xor(sacc[r], 2, 32);
        sacc[r] += __shfl_xor(sacc[r], 4, 32);
        sacc[r] += __shfl_xor(sacc[r], 8, 32);
    }

    if (lo < 8) {
        float sv = sacc[0];
#pragma unroll
        for (int r = 1; r < 8; ++r)
            if (lo == r) sv = sacc[r];
        int m = m0 + hi * 8 + lo;
        float sval = sv + b2[0];
        s_out[m] = sval;
        atomicMax(&segmax[index[m]], fenc(sval));
    }
}

// ---------------------------------------------- pass 2: exp + seg_sum
__global__ void pass2_k(float* __restrict__ s_e, const int* __restrict__ index,
                        const unsigned* __restrict__ segmax,
                        float* __restrict__ segsum, int n_msgs) {
    int i = blockIdx.x * blockDim.x + threadIdx.x;
    if (i >= n_msgs) return;
    int idx = index[i];
    float ev = __expf(s_e[i] - fdec(segmax[idx]));
    s_e[i] = ev;                       // in place: s -> e
    atomicAddF32(&segsum[idx], ev);
}

// ------------------------------------- pass 3: weighted segment scatter
// One wave per message: 128 floats = float4 per lane, atomic-add into out.
__global__ __launch_bounds__(256) void scatter_k(
    const float* __restrict__ msg, const int* __restrict__ index,
    const float* __restrict__ e, const float* __restrict__ segsum,
    float* __restrict__ out, int n_msgs) {
    int wave = threadIdx.x >> 5;
    int lane = threadIdx.x & 31;
    int m = blockIdx.x * 8 + wave;
    if (m >= n_msgs) return;
    int idx = index[m];
    float wgt = e[m] / (segsum[idx] + 1e-16f);
    float4 v = ((const float4*)(msg + (size_t)m * D))[lane];
    float* op = out + (size_t)idx * D + lane * 4;
    atomicAddF32(op + 0, wgt * v.x);
    atomicAddF32(op + 1, wgt * v.y);
    atomicAddF32(op + 2, wgt * v.z);
    atomicAddF32(op + 3, wgt * v.w);
}

extern "C" void kernel_launch(void* const* d_in, const int* in_sizes, int n_in,
                              void* d_out, int out_size, void* d_ws, size_t ws_size,
                              hipStream_t stream) {
    const float* msg   = (const float*)d_in[0];
    const int*   index = (const int*)d_in[1];
    // d_in[2] = t (unused), d_in[3] = dim_size (device scalar; derive from out_size)
    const float* W1 = (const float*)d_in[4];
    const float* b1 = (const float*)d_in[5];
    const float* W2 = (const float*)d_in[6];
    const float* b2 = (const float*)d_in[7];

    int n_msgs = in_sizes[0] / D;
    int dim    = out_size / D;

    char* ws = (char*)d_ws;
    float*     s_e    = (float*)ws;                                       // n_msgs f32
    float*     segsum = (float*)(ws + (size_t)n_msgs * 4);                // dim f32
    unsigned*  segmax = (unsigned*)(ws + (size_t)n_msgs * 4 + (size_t)dim * 4);
    _Float16*  bfrag  = (_Float16*)(ws + (size_t)n_msgs * 4 + (size_t)dim * 8); // 16KB

    float* out = (float*)d_out;

    init_k<<<(out_size + 255) / 256, 256, 0, stream>>>(out, segsum, segmax, out_size, dim);
    bprep_k<<<1, 512, 0, stream>>>(W1, bfrag);

    int waves  = (n_msgs + 15) / 16;
    int blocks = (waves + 7) / 8;
    score_k<<<blocks, 256, 0, stream>>>(msg, index, b1, W2, b2, bfrag,
                                        s_e, segmax, n_msgs);
    pass2_k<<<(n_msgs + 255) / 256, 256, 0, stream>>>(s_e, index, segmax, segsum, n_msgs);
    scatter_k<<<(n_msgs + 7) / 8, 256, 0, stream>>>(msg, index, s_e, segsum, out, n_msgs);
}
